// LinearAttention_20031727469014
// MI455X (gfx1250) — matrix operand order repaired and checked by reading, not compiled, not run on hardware
//
#include <hip/hip_runtime.h>
#include <hip/hip_bf16.h>

typedef __attribute__((ext_vector_type(16))) __bf16 v16bf;
typedef __attribute__((ext_vector_type(8)))  float  v8f;

union F8  { v8f  v; float f[8]; };
union U16 { v16bf v; unsigned int u[8]; };

#if __has_builtin(__builtin_amdgcn_tensor_load_to_lds)
#define USE_TDM 1
typedef __attribute__((ext_vector_type(4))) unsigned int tdm_g0_t;
typedef __attribute__((ext_vector_type(8))) int          tdm_g1_t;
typedef __attribute__((ext_vector_type(4))) int          tdm_g23_t;

// Issue one TDM copy: tile_d1 rows of tile_d0 dwords, global row stride
// stride0 dwords, into contiguous LDS at `l`. D# packing per ISA ch.8.
// amdgpu-toolchain (clang-23) 6-arg builtin form.
__device__ __forceinline__ void tdm_copy(const void* g, void* l,
                                         unsigned tile_d0, unsigned tile_d1,
                                         unsigned long long stride0) {
    unsigned long long ga  = (unsigned long long)(uintptr_t)g;
    unsigned           lds = (unsigned)(uintptr_t)l;  // low 32b of flat = LDS offset
    tdm_g0_t g0;
    g0[0] = 1u;                                   // count=1, user mode
    g0[1] = lds;                                  // lds_addr
    g0[2] = (unsigned)ga;                         // global_addr[31:0]
    g0[3] = (unsigned)(ga >> 32) | (2u << 30);    // global_addr[56:32] | type=2
    const unsigned long long td0 = 1ull << 20;    // tensor_dim0 (dwords), generous
    const unsigned long long td1 = 1ull << 20;    // tensor_dim1 (rows), generous
    tdm_g1_t g1;
    g1[0] = (int)(2u << 16);                      // wg_mask=0, data_size=2 (4B)
    g1[1] = (int)((td0 & 0xFFFFu) << 16);         // tensor_dim0[15:0] @bit48
    g1[2] = (int)((unsigned)(td0 >> 16) | ((unsigned)(td1 & 0xFFFFu) << 16));
    g1[3] = (int)(((unsigned)(td1 >> 16) & 0xFFFFu) | (tile_d0 << 16)); // tile_dim0
    g1[4] = (int)(tile_d1 & 0xFFFFu);             // tile_dim1 (tile_dim2=0)
    g1[5] = (int)(stride0 & 0xFFFFFFFFu);         // tensor_dim0_stride[31:0]
    g1[6] = (int)((unsigned)(stride0 >> 32) & 0xFFFFu); // stride[47:32], d1stride=0
    g1[7] = 0;
    tdm_g23_t z4 = {0, 0, 0, 0};
    tdm_g1_t  z8 = {0, 0, 0, 0, 0, 0, 0, 0};
    __builtin_amdgcn_tensor_load_to_lds(g0, g1, z4, z4, z8, 0);
}
#else
#define USE_TDM 0
#endif

// ---------------------------------------------------------------------------
// Kernel 1: RoPE(Q), RoPE(K) -> bf16 row-major; V -> bf16 transposed [b][d][s]
// ---------------------------------------------------------------------------
__global__ void rope_cast_kernel(const float* __restrict__ Q,
                                 const float* __restrict__ K,
                                 const float* __restrict__ V,
                                 __hip_bfloat16* __restrict__ Qbf,
                                 __hip_bfloat16* __restrict__ Kbf,
                                 __hip_bfloat16* __restrict__ Vt) {
    constexpr int S = 4096;
    constexpr int D = 128;
    const int bs = blockIdx.x;          // b*S + s
    const int s  = bs & (S - 1);
    const int b  = bs >> 12;
    const int d  = threadIdx.x;         // 0..63

    const float LN1E4 = 9.210340371976184f;   // ln(10000)
    float invf = __expf(-LN1E4 * (float)d * (1.0f / 64.0f));
    float ang  = (float)s * invf;
    float cs, sn;
    __sincosf(ang, &cs, &sn);

    size_t base = (size_t)bs * D;

    float q0 = Q[base + d], q1 = Q[base + d + 64];
    Qbf[base + d]      = __float2bfloat16(q0 * cs - q1 * sn);
    Qbf[base + d + 64] = __float2bfloat16(q1 * cs + q0 * sn);

    float k0 = K[base + d], k1 = K[base + d + 64];
    Kbf[base + d]      = __float2bfloat16(k0 * cs - k1 * sn);
    Kbf[base + d + 64] = __float2bfloat16(k1 * cs + k0 * sn);

    float v0 = V[base + d], v1 = V[base + d + 64];
    Vt[((size_t)b * D + d)      * S + s] = __float2bfloat16(v0);
    Vt[((size_t)b * D + d + 64) * S + s] = __float2bfloat16(v1);
}

// ---------------------------------------------------------------------------
// Per-tile compute body (scalar-branched at call site; EXEC all-ones here).
// ---------------------------------------------------------------------------
template <bool ACT0>
__device__ __forceinline__ void compute_tile(
    const __hip_bfloat16* __restrict__ Kt,   // LDS K tile  [32 keys][128 d]
    const __hip_bfloat16* __restrict__ Vc,   // LDS Vt tile [128 d][32 keys]
    __hip_bfloat16* __restrict__ sw0,
    __hip_bfloat16* __restrict__ sw1,
    const U16 (&qa)[2][4], F8 (&oC)[2][8],
    int s0, int qw, int lm, int g) {
    F8 sC0[2], sC1[2];
#pragma unroll
    for (int kt = 0; kt < 2; ++kt) {
        sC0[kt].v = (v8f){0.f, 0.f, 0.f, 0.f, 0.f, 0.f, 0.f, 0.f};
        sC1[kt].v = (v8f){0.f, 0.f, 0.f, 0.f, 0.f, 0.f, 0.f, 0.f};
    }

    // ---- S = Qr . Kr^T : kb fragment shared by both M-subtiles ----
#pragma unroll
    for (int kt = 0; kt < 2; ++kt) {
#pragma unroll
        for (int c = 0; c < 4; ++c) {
            U16 kb;
#pragma unroll
            for (int v = 0; v < 8; ++v) {
                int d   = 32 * c + 16 * (v >> 2) + 2 * (v & 3) + 8 * g;
                int key = kt * 16 + lm;
                kb.u[v] = *reinterpret_cast<const unsigned int*>(Kt + key * 128 + d);
            }
            if constexpr (ACT0)
                sC0[kt].v = __builtin_amdgcn_wmma_f32_16x16x32_bf16(
                    false, qa[0][c].v, false, kb.v, (short)0, sC0[kt].v, false, false);
            sC1[kt].v = __builtin_amdgcn_wmma_f32_16x16x32_bf16(
                false, qa[1][c].v, false, kb.v, (short)0, sC1[kt].v, false, false);
        }
    }

    // ---- causal mask + re-stripe through per-wave LDS scratch ----
    if constexpr (ACT0) {
        if (s0 + 31 > qw) {
#pragma unroll
            for (int kt = 0; kt < 2; ++kt) {
                int key = s0 + kt * 16 + lm;
#pragma unroll
                for (int r = 0; r < 8; ++r)
                    if (key > qw + r + 8 * g) sC0[kt].f[r] = 0.f;
            }
        }
#pragma unroll
        for (int kt = 0; kt < 2; ++kt)
#pragma unroll
            for (int r = 0; r < 8; ++r)
                sw0[(r + 8 * g) * 32 + kt * 16 + lm] = __float2bfloat16(sC0[kt].f[r]);
    }
    {
        const int qm = qw + 16;
        if (s0 + 31 > qm) {
#pragma unroll
            for (int kt = 0; kt < 2; ++kt) {
                int key = s0 + kt * 16 + lm;
#pragma unroll
                for (int r = 0; r < 8; ++r)
                    if (key > qm + r + 8 * g) sC1[kt].f[r] = 0.f;
            }
        }
#pragma unroll
        for (int kt = 0; kt < 2; ++kt)
#pragma unroll
            for (int r = 0; r < 8; ++r)
                sw1[(r + 8 * g) * 32 + kt * 16 + lm] = __float2bfloat16(sC1[kt].f[r]);
    }

    // ---- read S back in A-fragment layout ----
    U16 sa0, sa1;
#pragma unroll
    for (int p = 0; p < 8; ++p) {
        int k = 16 * (p >> 2) + 2 * (p & 3) + 8 * g;
        if constexpr (ACT0)
            sa0.u[p] = *reinterpret_cast<const unsigned int*>(sw0 + lm * 32 + k);
        sa1.u[p] = *reinterpret_cast<const unsigned int*>(sw1 + lm * 32 + k);
    }

    // ---- O += S . V : vb fragment shared by both M-subtiles ----
#pragma unroll
    for (int c = 0; c < 8; ++c) {
        U16 vb;
#pragma unroll
        for (int v = 0; v < 8; ++v) {
            int key = 16 * (v >> 2) + 2 * (v & 3) + 8 * g;
            int d   = c * 16 + lm;
            vb.u[v] = *reinterpret_cast<const unsigned int*>(Vc + d * 32 + key);
        }
        if constexpr (ACT0)
            oC[0][c].v = __builtin_amdgcn_wmma_f32_16x16x32_bf16(
                false, sa0.v, false, vb.v, (short)0, oC[0][c].v, false, false);
        oC[1][c].v = __builtin_amdgcn_wmma_f32_16x16x32_bf16(
            false, sa1.v, false, vb.v, (short)0, oC[1][c].v, false, false);
    }
}

// ---------------------------------------------------------------------------
// Kernel 2: streaming causal  out = tril(Qr Kr^T) V   (bf16 WMMA, f32 acc)
// 128 threads = 4 wave32, 32 queries x 128 D per wave. 32-key tiles stream
// through double-buffered LDS filled by the Tensor Data Mover (wave 0 issues,
// s_wait_tensorcnt + block barrier publishes).
// ---------------------------------------------------------------------------
__global__ __launch_bounds__(128) void attn_kernel(
    const __hip_bfloat16* __restrict__ Qbf,
    const __hip_bfloat16* __restrict__ Kbf,
    const __hip_bfloat16* __restrict__ Vt,
    float* __restrict__ out) {
    constexpr int S = 4096;
    constexpr int D = 128;

    __shared__ __align__(16) __hip_bfloat16 Kl[2][32 * 128];   // [key][d]
    __shared__ __align__(16) __hip_bfloat16 Vl[2][128 * 32];   // [d][key]
    __shared__ __align__(16) __hip_bfloat16 Sl[4][2][16 * 32]; // per-wave scratch

    const int t    = threadIdx.x;
    const int lane = t & 31;
    const int g    = lane >> 4;
    const int lm   = lane & 15;

    const int bb = blockIdx.y;
    const int q0 = blockIdx.x * 128;
    const int wu  = __builtin_amdgcn_readfirstlane(t >> 5);  // wave-uniform wave id
    const int qwu = q0 + 32 * wu;                            // first query row

    // ---- preload Q A-fragments for both 16-row M-subtiles ----
    U16 qa[2][4];
#pragma unroll
    for (int mt = 0; mt < 2; ++mt) {
        const __hip_bfloat16* qrow = Qbf + ((size_t)bb * S + qwu + 16 * mt + lm) * D;
#pragma unroll
        for (int c = 0; c < 4; ++c) {
#pragma unroll
            for (int p = 0; p < 8; ++p) {
                int k = 32 * c + 16 * (p >> 2) + 2 * (p & 3) + 8 * g;
                qa[mt][c].u[p] = *reinterpret_cast<const unsigned int*>(qrow + k);
            }
        }
    }

    F8 oC[2][8];
#pragma unroll
    for (int mt = 0; mt < 2; ++mt)
#pragma unroll
        for (int c = 0; c < 8; ++c)
            oC[mt][c].v = (v8f){0.f, 0.f, 0.f, 0.f, 0.f, 0.f, 0.f, 0.f};

    const int jmax = (q0 + 127) >> 5;

#if USE_TDM
    // ---- prologue: DMA tile 0 into buffer 0 ----
    if (wu == 0) {
        tdm_copy(Kbf + ((size_t)bb * S) * D, Kl[0], /*d0=*/2048, /*d1=*/1, 0);
        tdm_copy(Vt + (size_t)bb * D * S, Vl[0], /*d0=*/16, /*d1=*/128, /*stride=*/2048);
        __builtin_amdgcn_s_wait_tensorcnt(0);
    }
    __syncthreads();

    for (int j = 0; j <= jmax; ++j) {
        const int cur = j & 1;
        const int s0  = j * 32;
        if (j < jmax && wu == 0) {   // DMA next tile while everyone computes
            const int s1 = s0 + 32;
            tdm_copy(Kbf + ((size_t)bb * S + s1) * D, Kl[1 - cur], 2048, 1, 0);
            tdm_copy(Vt + (size_t)bb * D * S + s1, Vl[1 - cur], 16, 128, 2048);
        }

        if (s0 <= qwu + 31) {
            if (s0 <= qwu + 15)
                compute_tile<true>(Kl[cur], Vl[cur], Sl[wu][0], Sl[wu][1],
                                   qa, oC, s0, qwu, lm, g);
            else
                compute_tile<false>(Kl[cur], Vl[cur], Sl[wu][0], Sl[wu][1],
                                    qa, oC, s0, qwu, lm, g);
        }

        if (j < jmax && wu == 0) __builtin_amdgcn_s_wait_tensorcnt(0);
        __syncthreads();   // publishes next buffer; all reads of cur are done
    }
#else
    // ---- fallback: register-staged double buffering ----
    uint4 kst[4], vst[4];
    auto stage = [&](int j) {
        const int s0 = j * 32;
        const uint4* ksrc =
            reinterpret_cast<const uint4*>(Kbf + ((size_t)bb * S + s0) * D);
        kst[0] = ksrc[t];
        kst[1] = ksrc[t + 128];
        kst[2] = ksrc[t + 256];
        kst[3] = ksrc[t + 384];
        const uint4* vsrc =
            reinterpret_cast<const uint4*>(Vt + ((size_t)bb * D + t) * S + s0);
        vst[0] = vsrc[0];
        vst[1] = vsrc[1];
        vst[2] = vsrc[2];
        vst[3] = vsrc[3];
    };
    auto commit = [&](int buf) {
        uint4* kd = reinterpret_cast<uint4*>(Kl[buf]);
        kd[t]       = kst[0];
        kd[t + 128] = kst[1];
        kd[t + 256] = kst[2];
        kd[t + 384] = kst[3];
        uint4* vd = reinterpret_cast<uint4*>(Vl[buf]) + t * 4;
        vd[0] = vst[0];
        vd[1] = vst[1];
        vd[2] = vst[2];
        vd[3] = vst[3];
    };

    stage(0);
    commit(0);
    __syncthreads();

    for (int j = 0; j <= jmax; ++j) {
        const int cur = j & 1;
        const int s0  = j * 32;
        if (j < jmax) stage(j + 1);

        if (s0 <= qwu + 31) {
            if (s0 <= qwu + 15)
                compute_tile<true>(Kl[cur], Vl[cur], Sl[wu][0], Sl[wu][1],
                                   qa, oC, s0, qwu, lm, g);
            else
                compute_tile<false>(Kl[cur], Vl[cur], Sl[wu][0], Sl[wu][1],
                                    qa, oC, s0, qwu, lm, g);
        }

        if (j < jmax) commit(1 - cur);
        __syncthreads();
    }
#endif

    // ---- write 32x128 f32 output (C layout: row = r+8g, col = c*16+lm) ----
#pragma unroll
    for (int mt = 0; mt < 2; ++mt)
#pragma unroll
        for (int c = 0; c < 8; ++c)
#pragma unroll
            for (int r = 0; r < 8; ++r)
                out[((size_t)bb * S + qwu + 16 * mt + r + 8 * g) * D + c * 16 + lm] =
                    oC[mt][c].f[r];
}

// ---------------------------------------------------------------------------
extern "C" void kernel_launch(void* const* d_in, const int* in_sizes, int n_in,
                              void* d_out, int out_size, void* d_ws, size_t ws_size,
                              hipStream_t stream) {
    const float* Q = (const float*)d_in[0];
    const float* K = (const float*)d_in[1];
    const float* V = (const float*)d_in[2];
    float* out = (float*)d_out;

    const int S = 4096, D = 128;
    const int B = in_sizes[0] / (S * D);
    const size_t N = (size_t)B * S * D;

    __hip_bfloat16* Qbf  = (__hip_bfloat16*)d_ws;
    __hip_bfloat16* Kbf  = Qbf + N;
    __hip_bfloat16* Vtbf = Kbf + N;

    rope_cast_kernel<<<dim3(B * S), 64, 0, stream>>>(Q, K, V, Qbf, Kbf, Vtbf);
    attn_kernel<<<dim3(S / 128, B), 128, 0, stream>>>(Qbf, Kbf, Vtbf, out);
}
